// Seq2Seq_23639499997324
// MI455X (gfx1250) — matrix operand order repaired and checked by reading, compiled, not run on hardware
//
#include <hip/hip_runtime.h>
#include <math.h>

// ---------------------------------------------------------------------------
// MI455X (gfx1250) seq2seq: encoder bi-LSTM + attention decoder + classifier.
// All heavy GEMMs use v_wmma_f32_16x16x32_f16 (wave32). Fragment loads are
// vectorized (2x b128 per fragment) by storing the B operand K-innermost
// (transposed) in LDS / global. Recurrences run as persistent 1024-thread
// (32-wave) workgroups with h (f16) / c (f32) in the 320KB WGP LDS.
// N is always a multiple of 64 (2048 / 512 / 32000) -> no column guards.
// ---------------------------------------------------------------------------

typedef _Float16 half_t;
typedef __attribute__((ext_vector_type(16))) _Float16 v16h;
typedef __attribute__((ext_vector_type(8)))  _Float16 v8h;
typedef __attribute__((ext_vector_type(8)))  float    v8f;

#define Bn 32
#define Sn 128
#define Tn 64
#define Hn 512
#define VTn 32000
#define G4H 2048   // 4*H
#define H2 1024    // 2*H
#define T1 63      // T-1

// ---------------- WMMA fragment helpers (ISA 7.12.2 layouts) ----------------

// A matrix 16x32 f16 (row-major, ld=lda): lane L holds row M = row0 + (L&15);
// K base kb = k0 + (L>=16)*8; slots 0..7 -> kb..kb+7, slots 8..15 -> kb+16..kb+23.
__device__ __forceinline__ v16h load_a_frag(const half_t* A, int lda,
                                            int row0, int k0, int lane) {
  int m  = row0 + (lane & 15);
  int kb = k0 + ((lane >> 4) << 3);
  const half_t* p = A + (size_t)m * lda + kb;
  v8h lo = *(const v8h*)p;
  v8h hi = *(const v8h*)(p + 16);
  v16h a;
#pragma unroll
  for (int j = 0; j < 8; ++j) { a[j] = lo[j]; a[8 + j] = hi[j]; }
  return a;
}

// B matrix 32x16 f16 stored TRANSPOSED (K innermost): Bt[n][k], ld=ldk.
// Lane L holds col N = col0 + (L&15); kb = k0 + (L>=16)*8; slots 0..7 -> kb..kb+7, slots 8..15 -> kb+16..kb+23.
__device__ __forceinline__ v16h load_bt_frag(const half_t* Bt, int ldk,
                                             int k0, int col0, int lane) {
  int n  = col0 + (lane & 15);
  int kb = k0 + ((lane >> 4) << 3);
  const half_t* p = Bt + (size_t)n * ldk + kb;
  v8h lo = *(const v8h*)p;
  v8h hi = *(const v8h*)(p + 16);
  v16h b;
#pragma unroll
  for (int j = 0; j < 8; ++j) { b[j] = lo[j]; b[8 + j] = hi[j]; }
  return b;
}

__device__ __forceinline__ v8f wmma_f16(v16h a, v16h b, v8f c) {
  return __builtin_amdgcn_wmma_f32_16x16x32_f16(false, a, false, b,
                                                (short)0, c, false, false);
}

__device__ __forceinline__ float sigm(float x) { return 1.f / (1.f + __expf(-x)); }

// ---------------- small utility kernels ----------------

__global__ void embed_src_k(const int* __restrict__ src,
                            const float* __restrict__ E,
                            half_t* __restrict__ xs) {
  int idx = blockIdx.x * 256 + threadIdx.x;          // [s][b][h]
  if (idx >= Sn * Bn * Hn) return;
  int h = idx & (Hn - 1);
  int b = (idx >> 9) & (Bn - 1);
  int s = idx >> 14;
  int tok = src[b * Sn + s];
  xs[idx] = (half_t)E[(size_t)tok * Hn + h];
}

__global__ void embed_trg_k(const int* __restrict__ trg,
                            const float* __restrict__ E,
                            half_t* __restrict__ ys) {
  int idx = blockIdx.x * 256 + threadIdx.x;          // [t][b][h], t<63
  if (idx >= T1 * Bn * Hn) return;
  int h = idx & (Hn - 1);
  int b = (idx >> 9) & (Bn - 1);
  int t = idx >> 14;
  int tok = trg[b * Tn + t];
  ys[idx] = (half_t)E[(size_t)tok * Hn + h];
}

// U [H][4H] f32 -> Ut [4H][H] f16 (transposed, K innermost for B fragments).
__global__ void u_to_f16t_k(const float* __restrict__ in,
                            half_t* __restrict__ out) {
  int i = blockIdx.x * 256 + threadIdx.x;   // out-order: i = n*512 + k
  if (i >= Hn * G4H) return;
  int n = i >> 9;
  int k = i & (Hn - 1);
  out[i] = (half_t)in[(size_t)k * G4H + n];
}

// ---------------- generic WMMA GEMM: C(f32) = A(f16) x B(f32->f16) + bias ---
// Block = 256 threads (8 waves), 128x64 output tile, BK=32.
// Each wave computes a 32x32 patch = 2x2 WMMA tiles (4 WMMAs per k-step).
// B tile staged transposed (K innermost): thread owns one n column, 8
// consecutive k -> coalesced global reads + one ds_store_b128.
// permTB: row r = t*32+b is written to output row b*63+t (classifier fusion).
// Requires: N % 64 == 0, K % 32 == 0 (true for all call sites).

#define BM 128
#define BN 64
#define BK 32

__global__ __launch_bounds__(256)
void gemm_f16w_k(const half_t* __restrict__ A, int lda,
                 const float* __restrict__ Bw, int ldb,
                 const float* __restrict__ bias,
                 float* __restrict__ C, int ldc,
                 int M, int N, int K, int permTB) {
  __shared__ half_t As[BM * BK];   // [128][32] row-major, 8KB
  __shared__ half_t Bt[BN * BK];   // [64][32] transposed, 4KB
  int tid = threadIdx.x;
  int wave = tid >> 5, lane = tid & 31;
  int rowBase = blockIdx.y * BM;
  int colBase = blockIdx.x * BN;
  int m0 = (wave & 3) * 32;    // m-tiles m0, m0+16
  int n0 = (wave >> 2) * 32;   // n-tiles n0, n0+16

  // staging coordinates (loop-invariant)
  int aRow = tid >> 1;                  // A: thread copies 16 halfs of row aRow
  int aK   = (tid & 1) * 16;
  int bN   = tid & 63;                  // B: thread owns column bN,
  int bK0  = (tid >> 6) * 8;            //    k-range [bK0, bK0+8)

  v8f acc[2][2];
  acc[0][0] = (v8f){}; acc[0][1] = (v8f){};
  acc[1][0] = (v8f){}; acc[1][1] = (v8f){};

  for (int k0 = 0; k0 < K; k0 += BK) {
    // stage A: 16 contiguous halfs per thread via 2x b128 (zero-pad OOB rows)
    {
      int gr = rowBase + aRow;
      v8h z = {};
      const half_t* srcp = A + (size_t)gr * lda + k0 + aK;
      v8h lo = (gr < M) ? *(const v8h*)srcp       : z;
      v8h hi = (gr < M) ? *(const v8h*)(srcp + 8) : z;
      *(v8h*)&As[aRow * BK + aK]     = lo;
      *(v8h*)&As[aRow * BK + aK + 8] = hi;
    }
    // stage B transposed: 8 coalesced f32 loads -> cvt -> one ds_store_b128
    {
      const float* bp = Bw + (size_t)(k0 + bK0) * ldb + colBase + bN;
      v8h tmp;
#pragma unroll
      for (int j = 0; j < 8; ++j) tmp[j] = (half_t)bp[(size_t)j * ldb];
      *(v8h*)&Bt[bN * BK + bK0] = tmp;
    }
    __syncthreads();

    v16h a0 = load_a_frag(As, BK, m0, 0, lane);
    v16h a1 = load_a_frag(As, BK, m0 + 16, 0, lane);
    v16h b0 = load_bt_frag(Bt, BK, 0, n0, lane);
    v16h b1 = load_bt_frag(Bt, BK, 0, n0 + 16, lane);
    acc[0][0] = wmma_f16(a0, b0, acc[0][0]);
    acc[0][1] = wmma_f16(a0, b1, acc[0][1]);
    acc[1][0] = wmma_f16(a1, b0, acc[1][0]);
    acc[1][1] = wmma_f16(a1, b1, acc[1][1]);
    __syncthreads();

    if (k0 + BK < K) {       // lowers to global_prefetch_b8
      __builtin_prefetch(&A[(size_t)(rowBase + aRow) * lda + k0 + BK], 0, 1);
      __builtin_prefetch(&Bw[(size_t)(k0 + BK + bK0) * ldb + colBase + bN], 0, 1);
    }
  }

  int nIn = lane & 15;
  int mh  = (lane >> 4) * 8;
#pragma unroll
  for (int mi = 0; mi < 2; ++mi) {
#pragma unroll
    for (int r = 0; r < 8; ++r) {
      int gr = rowBase + m0 + mi * 16 + mh + r;
      if (gr >= M) continue;
      size_t outRow = permTB ? (size_t)((gr & 31) * T1 + (gr >> 5)) : (size_t)gr;
#pragma unroll
      for (int ni = 0; ni < 2; ++ni) {
        int gc = colBase + n0 + ni * 16 + nIn;
        C[outRow * (size_t)ldc + gc] = acc[mi][ni][r] + bias[gc];
      }
    }
  }
}

// ---------------- persistent encoder LSTM (one direction per launch) --------
// 1024 threads = 32 waves. Wave w owns gate-columns [w*16, w*16+16) of each of
// the 4 gate slices. h (f16, 32KB) and c (f32, 64KB) live in LDS.
// Ut is [4H][H] f16 (K innermost) -> B fragments are 2x global b128 per lane.

__global__ __launch_bounds__(1024)
void lstm_enc_k(const half_t* __restrict__ Ut,      // [4H][H] f16 transposed
                const float* __restrict__ xW,       // [S][B][4H] (x@W + bias)
                half_t* __restrict__ hs_cat,        // [S][B][2H]
                half_t* __restrict__ prevA,         // [64][2H]
                int backward, int colOff) {
  extern __shared__ char smem[];
  half_t* hls = (half_t*)smem;                       // [32][512]
  float*  cls = (float*)(smem + Bn * Hn * 2);        // [32][512]
  int tid = threadIdx.x;
  int wave = tid >> 5, lane = tid & 31;
  int n0 = wave * 16;

  for (int i = tid; i < Bn * Hn; i += 1024) { hls[i] = (half_t)0.f; cls[i] = 0.f; }
  __syncthreads();

  for (int step = 0; step < Sn; ++step) {
    int s = backward ? (Sn - 1 - step) : step;

    v8f acc[4][2];
#pragma unroll
    for (int g = 0; g < 4; ++g) { acc[g][0] = (v8f){}; acc[g][1] = (v8f){}; }

    for (int k0 = 0; k0 < Hn; k0 += 32) {
      v16h a0 = load_a_frag(hls, Hn, 0, k0, lane);
      v16h a1 = load_a_frag(hls, Hn, 16, k0, lane);
#pragma unroll
      for (int g = 0; g < 4; ++g) {
        v16h bf = load_bt_frag(Ut, Hn, k0, g * Hn + n0, lane);
        acc[g][0] = wmma_f16(a0, bf, acc[g][0]);
        acc[g][1] = wmma_f16(a1, bf, acc[g][1]);
      }
    }
    __syncthreads();   // all waves finished reading h of step-1

    const float* xWs = xW + (size_t)s * Bn * G4H;
    int n  = n0 + (lane & 15);
    int mh = (lane >> 4) * 8;
#pragma unroll
    for (int mt = 0; mt < 2; ++mt) {
#pragma unroll
      for (int r = 0; r < 8; ++r) {
        int b = mt * 16 + mh + r;
        const float* xr = xWs + (size_t)b * G4H;
        float zi = acc[0][mt][r] + xr[0 * Hn + n];
        float zf = acc[1][mt][r] + xr[1 * Hn + n];
        float zg = acc[2][mt][r] + xr[2 * Hn + n];
        float zo = acc[3][mt][r] + xr[3 * Hn + n];
        float cN = sigm(zf) * cls[b * Hn + n] + sigm(zi) * tanhf(zg);
        float h  = sigm(zo) * tanhf(cN);
        cls[b * Hn + n] = cN;
        hls[b * Hn + n] = (half_t)h;
        hs_cat[((size_t)s * Bn + b) * H2 + colOff + n] = (half_t)h;
      }
    }
    __syncthreads();
  }

  for (int i = tid; i < Bn * Hn; i += 1024) {
    int b = i >> 9, n = i & (Hn - 1);
    prevA[(size_t)b * H2 + colOff + n]        = hls[i];
    prevA[(size_t)(32 + b) * H2 + colOff + n] = (half_t)cls[i];
  }
}

// ---------------- persistent decoder: LSTM cell + attention -----------------

__global__ __launch_bounds__(1024)
void decoder_k(const half_t* __restrict__ Ut,     // Ud [4H][H] f16 transposed
               const float* __restrict__ yW,      // [63][B][4H] (y@Wd + bd)
               const float* __restrict__ prevHC,  // [64][H] f32
               const float* __restrict__ src_h,   // [S][B][H] f32
               const int* __restrict__ src,       // [B][S]
               half_t* __restrict__ dec16) {      // [63][B][2H] f16
  extern __shared__ char smem[];
  half_t* hls  = (half_t*)smem;                         // 32KB
  float*  cls  = (float*)(smem + Bn * Hn * 2);          // 64KB
  float*  attn = (float*)(smem + Bn * Hn * 2 + Bn * Hn * 4);  // [32][128] 16KB
  int tid = threadIdx.x;
  int wave = tid >> 5, lane = tid & 31;
  int n0 = wave * 16;

  for (int i = tid; i < Bn * Hn; i += 1024) {
    hls[i] = (half_t)prevHC[i];
    cls[i] = prevHC[Bn * Hn + i];
  }
  __syncthreads();

  for (int t = 0; t < T1; ++t) {
    // ---- LSTM cell: h@Ud via WMMA ----
    v8f acc[4][2];
#pragma unroll
    for (int g = 0; g < 4; ++g) { acc[g][0] = (v8f){}; acc[g][1] = (v8f){}; }

    for (int k0 = 0; k0 < Hn; k0 += 32) {
      v16h a0 = load_a_frag(hls, Hn, 0, k0, lane);
      v16h a1 = load_a_frag(hls, Hn, 16, k0, lane);
#pragma unroll
      for (int g = 0; g < 4; ++g) {
        v16h bf = load_bt_frag(Ut, Hn, k0, g * Hn + n0, lane);
        acc[g][0] = wmma_f16(a0, bf, acc[g][0]);
        acc[g][1] = wmma_f16(a1, bf, acc[g][1]);
      }
    }
    __syncthreads();

    const float* yWs = yW + (size_t)t * Bn * G4H;
    int n  = n0 + (lane & 15);
    int mh = (lane >> 4) * 8;
#pragma unroll
    for (int mt = 0; mt < 2; ++mt) {
#pragma unroll
      for (int r = 0; r < 8; ++r) {
        int b = mt * 16 + mh + r;
        const float* yr = yWs + (size_t)b * G4H;
        float zi = acc[0][mt][r] + yr[0 * Hn + n];
        float zf = acc[1][mt][r] + yr[1 * Hn + n];
        float zg = acc[2][mt][r] + yr[2 * Hn + n];
        float zo = acc[3][mt][r] + yr[3 * Hn + n];
        float cN = sigm(zf) * cls[b * Hn + n] + sigm(zi) * tanhf(zg);
        float h  = sigm(zo) * tanhf(cN);
        cls[b * Hn + n] = cN;
        hls[b * Hn + n] = (half_t)h;
        dec16[((size_t)t * Bn + b) * H2 + Hn + n] = (half_t)h;  // h half
      }
    }
    __syncthreads();

    // ---- attention: wave w handles batch b=w ----
    int b = wave;
    for (int sp = 0; sp < Sn; ++sp) {
      const float* sh = src_h + ((size_t)sp * Bn + b) * Hn;
      float p = 0.f;
      for (int hh = lane; hh < Hn; hh += 32)
        p += (float)hls[b * Hn + hh] * sh[hh];
#pragma unroll
      for (int off = 16; off > 0; off >>= 1) p += __shfl_xor(p, off, 32);
      if (lane == 0) {
        if (src[b * Sn + sp] == 0) p = -1e9f;
        attn[b * Sn + sp] = p;
      }
    }
    __syncthreads();

    // softmax over s (4 values per lane)
    float v[4];
    float mx = -3.4e38f;
#pragma unroll
    for (int j = 0; j < 4; ++j) {
      v[j] = attn[b * Sn + lane + j * 32];
      mx = fmaxf(mx, v[j]);
    }
#pragma unroll
    for (int off = 16; off > 0; off >>= 1) mx = fmaxf(mx, __shfl_xor(mx, off, 32));
    float sum = 0.f;
#pragma unroll
    for (int j = 0; j < 4; ++j) { v[j] = __expf(v[j] - mx); sum += v[j]; }
#pragma unroll
    for (int off = 16; off > 0; off >>= 1) sum += __shfl_xor(sum, off, 32);
    float inv = 1.f / sum;
#pragma unroll
    for (int j = 0; j < 4; ++j) attn[b * Sn + lane + j * 32] = v[j] * inv;
    __syncthreads();

    // weighted sum: ws[b][h] = sum_s attn[b][s] * src_h[s][b][h]
    for (int hh = lane; hh < Hn; hh += 32) {
      float w = 0.f;
      for (int sp = 0; sp < Sn; ++sp)
        w += attn[b * Sn + sp] * src_h[((size_t)sp * Bn + b) * Hn + hh];
      dec16[((size_t)t * Bn + b) * H2 + hh] = (half_t)w;   // ws half
    }
    __syncthreads();
  }
}

// ---------------------------------------------------------------------------

extern "C" void kernel_launch(void* const* d_in, const int* in_sizes, int n_in,
                              void* d_out, int out_size, void* d_ws, size_t ws_size,
                              hipStream_t stream) {
  const int*   src   = (const int*)d_in[0];
  const int*   trg   = (const int*)d_in[1];
  const float* E_src = (const float*)d_in[2];
  const float* E_trg = (const float*)d_in[3];
  const float* Wf = (const float*)d_in[4];
  const float* Uf = (const float*)d_in[5];
  const float* bf = (const float*)d_in[6];
  const float* Wb = (const float*)d_in[7];
  const float* Ub = (const float*)d_in[8];
  const float* bb = (const float*)d_in[9];
  const float* Wd = (const float*)d_in[10];
  const float* Ud = (const float*)d_in[11];
  const float* bd = (const float*)d_in[12];
  const float* fcW = (const float*)d_in[13];
  const float* fcb = (const float*)d_in[14];
  const float* clsW = (const float*)d_in[15];
  const float* clsb = (const float*)d_in[16];
  float* out = (float*)d_out;   // [B][63][VT]

  // workspace carve-up
  char* base = (char*)d_ws;
  size_t off = 0;
  auto carve = [&](size_t bytes) -> char* {
    char* p = base + off;
    off = (off + bytes + 255) & ~(size_t)255;
    return p;
  };
  half_t* xs16   = (half_t*)carve((size_t)Sn * Bn * Hn * 2);      // [S][B][H]
  half_t* ys16   = (half_t*)carve((size_t)T1 * Bn * Hn * 2);      // [63][B][H]
  half_t* Uf16   = (half_t*)carve((size_t)Hn * G4H * 2);          // [4H][H] transposed
  half_t* Ub16   = (half_t*)carve((size_t)Hn * G4H * 2);
  half_t* Ud16   = (half_t*)carve((size_t)Hn * G4H * 2);
  float*  xWf    = (float*)carve((size_t)Sn * Bn * G4H * 4);      // [S][B][4H]
  float*  xWb    = (float*)carve((size_t)Sn * Bn * G4H * 4);
  float*  yWd    = (float*)carve((size_t)T1 * Bn * G4H * 4);      // [63][B][4H]
  half_t* hs_cat = (half_t*)carve((size_t)Sn * Bn * H2 * 2);      // [S][B][2H]
  float*  src_h  = (float*)carve((size_t)Sn * Bn * Hn * 4);       // [S][B][H]
  half_t* prevA  = (half_t*)carve((size_t)64 * H2 * 2);           // [64][2H]
  float*  prevHC = (float*)carve((size_t)64 * Hn * 4);            // [64][H]
  half_t* dec16  = (half_t*)carve((size_t)T1 * Bn * H2 * 2);      // [63][B][2H]

  // 1. embeddings -> f16 activations
  embed_src_k<<<(Sn * Bn * Hn + 255) / 256, 256, 0, stream>>>(src, E_src, xs16);
  embed_trg_k<<<(T1 * Bn * Hn + 255) / 256, 256, 0, stream>>>(trg, E_trg, ys16);

  // 2. recurrent weights: f32 [H][4H] -> f16 transposed [4H][H] (once)
  int nU = Hn * G4H;
  u_to_f16t_k<<<(nU + 255) / 256, 256, 0, stream>>>(Uf, Uf16);
  u_to_f16t_k<<<(nU + 255) / 256, 256, 0, stream>>>(Ub, Ub16);
  u_to_f16t_k<<<(nU + 255) / 256, 256, 0, stream>>>(Ud, Ud16);

  // 3. hoisted input GEMMs: xW = xs@W + bias, yW = ys@Wd + bd
  {
    dim3 g(G4H / BN, (Sn * Bn) / BM);
    gemm_f16w_k<<<g, 256, 0, stream>>>(xs16, Hn, Wf, G4H, bf, xWf, G4H,
                                       Sn * Bn, G4H, Hn, 0);
    gemm_f16w_k<<<g, 256, 0, stream>>>(xs16, Hn, Wb, G4H, bb, xWb, G4H,
                                       Sn * Bn, G4H, Hn, 0);
  }
  {
    dim3 g(G4H / BN, (T1 * Bn + BM - 1) / BM);
    gemm_f16w_k<<<g, 256, 0, stream>>>(ys16, Hn, Wd, G4H, bd, yWd, G4H,
                                       T1 * Bn, G4H, Hn, 0);
  }

  // 4. persistent bi-LSTM encoder (h@U via WMMA, state in 96KB LDS)
  size_t encLds = (size_t)Bn * Hn * 2 + (size_t)Bn * Hn * 4;   // 96KB
  lstm_enc_k<<<1, 1024, encLds, stream>>>(Uf16, xWf, hs_cat, prevA, 0, 0);
  lstm_enc_k<<<1, 1024, encLds, stream>>>(Ub16, xWb, hs_cat, prevA, 1, Hn);

  // 5. src_h = hs_cat @ fcW + fcb ; prev_h/c = [h||c concat] @ fcW + fcb
  {
    dim3 g(Hn / BN, (Sn * Bn) / BM);
    gemm_f16w_k<<<g, 256, 0, stream>>>(hs_cat, H2, fcW, Hn, fcb, src_h, Hn,
                                       Sn * Bn, Hn, H2, 0);
  }
  {
    dim3 g(Hn / BN, 1);
    gemm_f16w_k<<<g, 256, 0, stream>>>(prevA, H2, fcW, Hn, fcb, prevHC, Hn,
                                       64, Hn, H2, 0);
  }

  // 6. persistent decoder: LSTM cell (WMMA) + attention, 112KB LDS
  size_t decLds = encLds + (size_t)Bn * Sn * 4;
  decoder_k<<<1, 1024, decLds, stream>>>(Ud16, yWd, prevHC, src_h, src, dec16);

  // 7. classifier: dec[2016,1024] @ clsW[1024,32000] + clsb -> out (permuted)
  {
    dim3 g(VTn / BN, (T1 * Bn + BM - 1) / BM);
    gemm_f16w_k<<<g, 256, 0, stream>>>(dec16, H2, clsW, VTn, clsb, out, VTn,
                                       T1 * Bn, VTn, H2, 1);
  }
}
